// ConvCls_85323820302553
// MI455X (gfx1250) — matrix operand order repaired and checked by reading, not compiled, not run on hardware
//
#include <hip/hip_runtime.h>

// CDNA5 WMMA vector types
typedef __attribute__((ext_vector_type(16))) __bf16 v16bf;
typedef __attribute__((ext_vector_type(8)))  float  v8f;

// fp32 -> packed 2x bf16 with round-to-nearest-even
static __device__ __forceinline__ unsigned int pack_bf16(float a, float b) {
    unsigned int ua = __builtin_bit_cast(unsigned int, a);
    unsigned int ub = __builtin_bit_cast(unsigned int, b);
    ua = ua + 0x7FFFu + ((ua >> 16) & 1u);
    ub = ub + 0x7FFFu + ((ub >> 16) & 1u);
    return (ua >> 16) | (ub & 0xFFFF0000u);
}

#define BB   512   // batch
#define CC   512   // channels (K)
#define UU   512   // unique labels (classifier subset rows)
#define HWD  49    // 7*7
#define NPAD 64    // padded spatial

// ---------------------------------------------------------------------------
// Kernel 1: rank-sort labels (all distinct by construction) + gather bias.
// ---------------------------------------------------------------------------
__global__ __launch_bounds__(512) void pfc_sort_labels(
    const int* __restrict__ labels, const float* __restrict__ bias,
    int* __restrict__ lu, float* __restrict__ bsub) {
    __shared__ int sl[BB];
    const int tid = threadIdx.x;
    const int myl = labels[tid];
    sl[tid] = myl;
    __syncthreads();
    int rank = 0;
    for (int j = 0; j < BB; ++j) {
        const int lj = sl[j];
        rank += (lj < myl) || (lj == myl && j < tid);
    }
    lu[rank]   = myl;
    bsub[rank] = bias[myl];
}

// ---------------------------------------------------------------------------
// Kernel 2: gather selected classifier rows, convert fp32 -> packed bf16.
// wbf layout: [u][c/2] as u32 (pair c, c+1). 512x256 u32 = 512 KB (L2-resident).
// ---------------------------------------------------------------------------
__global__ __launch_bounds__(256) void pfc_gather_w(
    const float* __restrict__ weight, const int* __restrict__ lu,
    unsigned int* __restrict__ wbf) {
    const int u   = blockIdx.x;
    const int t   = threadIdx.x;
    const int row = lu[u];
    const float* src = weight + (size_t)row * CC;
    const int c = t * 2;
    wbf[(size_t)u * (CC / 2) + t] = pack_bf16(src[c], src[c + 1]);
}

// ---------------------------------------------------------------------------
// Kernel 3: per-batch GEMM  out[b] = Wsub(bf16) x X_b(bf16) + bsub
//   One workgroup (8 waves) per batch image.
//   LDS holds X_b as bf16 pairs PRE-SWIZZLED into WMMA B-fragment order:
//     ldsX[(((kk*4 + nt)*16 + l15)*2 + sel)*8 + v]
//       = pack(x[K], x[K+1]),  K = kk*32 + sel*16 + 2v,  n = nt*16 + l15
//   so each B tile is two contiguous ds_load_b128 per lane (no reg shuffles).
//   Each wave: 64 U-rows x 64 N-cols = 4x4 tiles of v_wmma_f32_16x16x32_bf16.
// ---------------------------------------------------------------------------
__global__ __launch_bounds__(256) void pfc_gemm(
    const float* __restrict__ x, const unsigned int* __restrict__ wbf,
    const float* __restrict__ bsub, float* __restrict__ out) {
    __shared__ unsigned int ldsX[(CC / 2) * NPAD];  // 16384 u32 = 64 KB

    const int b   = blockIdx.x;
    const int tid = threadIdx.x;
    const size_t xbase = (size_t)b * CC * HWD;

    // Fill: source-coalesced global reads, swizzled LDS scatter.
    for (int i = tid; i < (CC / 2) * NPAD; i += 256) {
        const int k2 = i >> 6;          // K-pair index 0..255
        const int n  = i & (NPAD - 1);  // spatial col 0..63
        unsigned int p = 0u;
        if (n < HWD) {
            const float f0 = x[xbase + (size_t)(2 * k2)     * HWD + n];
            const float f1 = x[xbase + (size_t)(2 * k2 + 1) * HWD + n];
            p = pack_bf16(f0, f1);
        }
        const int kkf = k2 >> 4;         // K-step
        const int self = (k2 >> 3) & 1;  // half-wave select
        const int vf  = k2 & 7;          // fragment VGPR index
        const int ntf = n >> 4;          // n-tile
        const int lf  = n & 15;          // lane-in-half
        ldsX[((((kkf << 2) + ntf) * 16 + lf) * 2 + self) * 8 + vf] = p;
    }
    __syncthreads();

    const int wave  = tid >> 5;
    const int lane  = tid & 31;
    const int sel   = lane >> 4;
    const int l15   = lane & 15;
    const int ubase = wave * 64;   // 64 U-rows per wave

    v8f acc[4][4];
#pragma unroll
    for (int m = 0; m < 4; ++m)
#pragma unroll
        for (int n = 0; n < 4; ++n)
            acc[m][n] = (v8f){0.f, 0.f, 0.f, 0.f, 0.f, 0.f, 0.f, 0.f};

    // Per-lane byte-constant offsets
    const int fragoff = (l15 * 2 + sel) * 8;   // u32 offset within a (kk,nt) 256-u32 block

    union Frag { uint4 q[2]; v16bf v; };

    for (int kk = 0; kk < CC / 32; ++kk) {   // 16 K-steps of 32
        // A tiles: 16x32 bf16 per m-tile, two b128 global loads per lane.
        Frag A[4];
#pragma unroll
        for (int m = 0; m < 4; ++m) {
            const uint4* ap = (const uint4*)(wbf + (size_t)(ubase + m * 16 + l15) * (CC / 2));
            A[m].q[0] = ap[kk * 4 + sel * 2];
            A[m].q[1] = ap[kk * 4 + sel * 2 + 1];
        }
#pragma unroll
        for (int nt = 0; nt < 4; ++nt) {
            // B tile: two contiguous ds_load_b128 per lane, already in fragment order.
            const uint4* bp = (const uint4*)&ldsX[(kk * 4 + nt) * 256 + fragoff];
            Frag Bm;
            Bm.q[0] = bp[0];
            Bm.q[1] = bp[1];
#pragma unroll
            for (int m = 0; m < 4; ++m)
                acc[m][nt] = __builtin_amdgcn_wmma_f32_16x16x32_bf16(
                    false, A[m].v, false, Bm.v, (short)0, acc[m][nt],
                    false, false);
        }
    }

    // Epilogue: C/D layout — VGPR r, lane: M = r + sel*8, N = l15.
#pragma unroll
    for (int m = 0; m < 4; ++m) {
#pragma unroll
        for (int nt = 0; nt < 4; ++nt) {
            const int hw = nt * 16 + l15;
            if (hw < HWD) {
#pragma unroll
                for (int r = 0; r < 8; ++r) {
                    const int u = ubase + m * 16 + sel * 8 + r;
                    out[((size_t)b * UU + u) * HWD + hw] = acc[m][nt][r] + bsub[u];
                }
            }
        }
    }
}

// ---------------------------------------------------------------------------
extern "C" void kernel_launch(void* const* d_in, const int* in_sizes, int n_in,
                              void* d_out, int out_size, void* d_ws, size_t ws_size,
                              hipStream_t stream) {
    const float* x      = (const float*)d_in[0];   // [512,512,7,7]
    const int*   labels = (const int*)d_in[1];     // [512]
    const float* weight = (const float*)d_in[2];   // [85742,512,1,1]
    const float* bias   = (const float*)d_in[3];   // [85742]
    float* out = (float*)d_out;                    // [512,512,7,7]

    // Workspace: lu (2 KB) | bsub (2 KB) | wbf bf16 weights (512 KB)
    int*          lu   = (int*)d_ws;
    float*        bsub = (float*)((char*)d_ws + 2048);
    unsigned int* wbf  = (unsigned int*)((char*)d_ws + 4096);

    pfc_sort_labels<<<1, 512, 0, stream>>>(labels, bias, lu, bsub);
    pfc_gather_w  <<<UU, 256, 0, stream>>>(weight, lu, wbf);
    pfc_gemm      <<<BB, 256, 0, stream>>>(x, wbf, bsub, out);
}
